// MoE_MambaLayer_41798621725047
// MI455X (gfx1250) — compile-verified, matrix operand order checked by reading, not run on hardware
//
#include <hip/hip_runtime.h>

// ---- problem constants ------------------------------------------------------
#define T_TOK   8000   // B*L tokens
#define L_SEQ   2000
#define B_SZ    4
#define DM      512    // D_MODEL
#define DI      1024   // D_INNER
#define DSTATE  128
#define NH      16
#define HD      64
#define CDIM    1280   // CONV_DIM
#define DIN     2320   // 2*DI + 2*DSTATE + NH
#define NEXP    4

typedef __bf16 bf16_t;
typedef __attribute__((ext_vector_type(16))) __bf16 v16bf;
typedef __attribute__((ext_vector_type(8)))  __bf16 v8bf;
typedef __attribute__((ext_vector_type(8)))  float  v8f;

// ---- WMMA fragment helpers (CDNA5 16-bit A/B layout, wave32) ---------------
// A 16x32 (MxK): lanes 0-15 row M=lane, K 0..7 & 16..23 ; lanes 16-31 same row,
// K 8..15 & 24..31.  B 32x16 mirrors with N=lane&15 (W stored row-major [N][K]).
__device__ __forceinline__ v16bf load_frag(const bf16_t* __restrict__ base, int ld) {
  const int lane = threadIdx.x & 31;
  const int r  = lane & 15;
  const int kb = (lane >> 4) << 3;
  const bf16_t* p = base + (size_t)r * ld + kb;
  v8bf lo = *(const v8bf*)(p);
  v8bf hi = *(const v8bf*)(p + 16);
  v16bf f;
#pragma unroll
  for (int i = 0; i < 8; ++i) { f[i] = lo[i]; f[i + 8] = hi[i]; }
  return f;
}

__device__ __forceinline__ v8f wmma_bf16(const v16bf& a, const v16bf& b, const v8f& c) {
  return __builtin_amdgcn_wmma_f32_16x16x32_bf16(false, a, false, b, (short)0, c,
                                                 false, false);
}

// store one 16x16 f32 tile; `o` pre-offset to (m0, n0), row stride ldc
__device__ __forceinline__ void store_tile(float* __restrict__ o, int ldc, const v8f& c) {
  const int lane = threadIdx.x & 31;
  const int col  = lane & 15;
  const int rb   = (lane >> 4) << 3;
#pragma unroll
  for (int r = 0; r < 8; ++r) o[(size_t)(rb + r) * ldc + col] = c[r];
}

// per-wave 32(M) x NS*16(N) GEMM body, K compile-time, all indices static.
// Arow: A at (m0, 0), ld = K.  Brow: B(=weights row-major [N][K]) at (nbase, 0).
template <int NS, int K>
__device__ __forceinline__ void wave_gemm32(const bf16_t* __restrict__ Arow,
                                            const bf16_t* __restrict__ Brow,
                                            v8f (&acc)[2][4]) {
  for (int k0 = 0; k0 < K; k0 += 32) {
    v16bf a0 = load_frag(Arow + k0, K);
    v16bf a1 = load_frag(Arow + (size_t)16 * K + k0, K);
    if (k0 + 32 < K)
      __builtin_prefetch(Arow + k0 + 32, 0, 3);
#pragma unroll
    for (int s = 0; s < NS; ++s) {
      v16bf b = load_frag(Brow + (size_t)(s * 16) * K + k0, K);
      acc[0][s] = wmma_bf16(a0, b, acc[0][s]);
      acc[1][s] = wmma_bf16(a1, b, acc[1][s]);
    }
  }
}

// ---- gating: one wave per token --------------------------------------------
__global__ void k_gate(const float* __restrict__ x, const float* __restrict__ gw,
                       const float* __restrict__ gb, const float* __restrict__ noise,
                       float* __restrict__ maskw) {
  const int tok  = (blockIdx.x * blockDim.x + threadIdx.x) >> 5;
  const int lane = threadIdx.x & 31;
  if (tok >= T_TOK) return;
  const float* xr = x + (size_t)tok * DM;
  float lg[NEXP];
#pragma unroll
  for (int e = 0; e < NEXP; ++e) {
    float s = 0.f;
    for (int k = lane; k < DM; k += 32) s += xr[k] * gw[e * DM + k];
#pragma unroll
    for (int off = 16; off; off >>= 1) s += __shfl_xor(s, off, 32);
    lg[e] = s + gb[e] + 0.01f * noise[(size_t)tok * NEXP + e];
  }
  if (lane == 0) {
    float mx = fmaxf(fmaxf(lg[0], lg[1]), fmaxf(lg[2], lg[3]));
    float sc[NEXP]; float sum = 0.f;
#pragma unroll
    for (int e = 0; e < NEXP; ++e) { sc[e] = __expf(lg[e] - mx); sum += sc[e]; }
#pragma unroll
    for (int e = 0; e < NEXP; ++e) sc[e] /= sum;
    int i1 = 0;
    for (int e = 1; e < NEXP; ++e) if (sc[e] > sc[i1]) i1 = e;
    int i2 = -1;
    for (int e = 0; e < NEXP; ++e) if (e != i1 && (i2 < 0 || sc[e] > sc[i2])) i2 = e;
    const float denom = sc[i1] + sc[i2];
#pragma unroll
    for (int e = 0; e < NEXP; ++e)
      maskw[(size_t)tok * NEXP + e] = (e == i1 || e == i2) ? sc[e] / denom : 0.f;
  }
}

// ---- fp32 -> bf16 conversion ------------------------------------------------
__global__ void k_cvt(const float* __restrict__ in, bf16_t* __restrict__ out, int n) {
  int i = blockIdx.x * blockDim.x + threadIdx.x;
  if (i < n) out[i] = (bf16_t)in[i];
}

__global__ void k_zero(float* __restrict__ out, int n) {
  int i = blockIdx.x * blockDim.x + threadIdx.x;
  if (i < n) out[i] = 0.f;
}

// ---- in_proj GEMM: zx[T,2320] = Xb[T,512] * Wi^T ---------------------------
// block = 256 threads = 8 waves; each wave: 32(M) x 64(N), NS-templated edge.
__global__ void k_gemm_inproj(const bf16_t* __restrict__ Xb,
                              const bf16_t* __restrict__ Wb,
                              float* __restrict__ Cm) {
  const int wv    = threadIdx.x >> 5;
  const int m0    = blockIdx.y << 5;               // 32 rows per block-row
  const int nbase = (blockIdx.x << 9) + (wv << 6); // 8 waves * 64 cols
  int nvalid = (DIN - nbase) >> 4;
  if (nvalid > 4) nvalid = 4;
  if (nvalid <= 0) return;                          // wave-uniform
  v8f acc[2][4] = {};
  const bf16_t* Arow = Xb + (size_t)m0 * DM;
  const bf16_t* Brow = Wb + (size_t)nbase * DM;
  switch (nvalid) {                                 // wave-uniform dispatch
    case 4: wave_gemm32<4, DM>(Arow, Brow, acc); break;
    case 3: wave_gemm32<3, DM>(Arow, Brow, acc); break;
    case 2: wave_gemm32<2, DM>(Arow, Brow, acc); break;
    default: wave_gemm32<1, DM>(Arow, Brow, acc); break;
  }
#pragma unroll
  for (int s = 0; s < 4; ++s) {
    if (s < nvalid) {                               // uniform, static indices
      store_tile(Cm + (size_t)m0 * DIN + nbase + (s << 4), DIN, acc[0][s]);
      store_tile(Cm + (size_t)(m0 + 16) * DIN + nbase + (s << 4), DIN, acc[1][s]);
    }
  }
}

// ---- depthwise causal conv (D_CONV=4) + bias + SiLU ------------------------
__global__ void k_conv(const float* __restrict__ zx, const float* __restrict__ cw,
                       const float* __restrict__ cb, float* __restrict__ xc) {
  const int idx = blockIdx.x * blockDim.x + threadIdx.x;
  if (idx >= T_TOK * CDIM) return;
  const int c   = idx % CDIM;
  const int tok = idx / CDIM;
  const int b   = tok / L_SEQ;
  const int l   = tok % L_SEQ;
  float acc = cb[c];
#pragma unroll
  for (int j = 0; j < 4; ++j) {
    const int ls = l - 3 + j;
    if (ls >= 0)
      acc = fmaf(cw[c * 4 + j], zx[(size_t)(b * L_SEQ + ls) * DIN + DI + c], acc);
  }
  xc[(size_t)idx] = acc / (1.f + __expf(-acc));
}

// ---- sequential SSM scan: block per (batch, head); h[64x128] in registers --
__global__ void k_scan(const float* __restrict__ zx, const float* __restrict__ xc,
                       const float* __restrict__ dtb, const float* __restrict__ alog,
                       const float* __restrict__ Dv, float* __restrict__ yraw) {
  const int bh  = blockIdx.x;
  const int b   = bh >> 4;
  const int hh  = bh & 15;
  const int tid = threadIdx.x;
  const int p   = tid >> 2;           // head-dim element owned (0..63)
  const int nb  = (tid & 3) * 32;     // 32-state slice (0..127)
  __shared__ float xs[HD], Bs[DSTATE], Cs[DSTATE];
  __shared__ float sdt, sdA;
  const float A      = -__expf(alog[hh]);
  const float Dh     = Dv[hh];
  const float dtbias = dtb[hh];
  float h[32];
#pragma unroll
  for (int j = 0; j < 32; ++j) h[j] = 0.f;

  for (int l = 0; l < L_SEQ; ++l) {
    const int tok = b * L_SEQ + l;
    const float* xrow = xc + (size_t)tok * CDIM;
    if (tid < 128) Bs[tid] = xrow[DI + tid];
    else           Cs[tid - 128] = xrow[DI + DSTATE + (tid - 128)];
    if (tid < HD)  xs[tid] = xrow[hh * HD + tid];
    if (tid == 0) {
      const float raw = zx[(size_t)tok * DIN + (DI + CDIM) + hh] + dtbias;
      const float dt = (raw > 20.f) ? raw : log1pf(__expf(raw));
      sdt = dt;
      sdA = __expf(dt * A);
    }
    __syncthreads();
    const float xp   = xs[p];
    const float dtxp = sdt * xp;
    const float dA   = sdA;
    float part = 0.f;
#pragma unroll
    for (int j = 0; j < 32; ++j) {
      h[j] = fmaf(h[j], dA, dtxp * Bs[nb + j]);
      part = fmaf(h[j], Cs[nb + j], part);
    }
    part += __shfl_xor(part, 1, 32);
    part += __shfl_xor(part, 2, 32);
    if ((tid & 3) == 0)
      yraw[(size_t)tok * DI + hh * HD + p] = part + Dh * xp;
    __syncthreads();
  }
}

// ---- gated RMSNorm + bf16 cast: block per token ----------------------------
__global__ void k_norm(const float* __restrict__ yraw, const float* __restrict__ zx,
                       const float* __restrict__ nw, bf16_t* __restrict__ ybf) {
  const int tok = blockIdx.x;
  const int tid = threadIdx.x;
  __shared__ float red[256];
  float v[4];
  float ss = 0.f;
#pragma unroll
  for (int i = 0; i < 4; ++i) {
    const int d  = tid + i * 256;
    const float z = zx[(size_t)tok * DIN + d];           // z = first DI cols
    const float g = z / (1.f + __expf(-z));              // silu(z)
    const float val = yraw[(size_t)tok * DI + d] * g;
    v[i] = val;
    ss += val * val;
  }
  red[tid] = ss;
  __syncthreads();
  for (int s = 128; s; s >>= 1) {
    if (tid < s) red[tid] += red[tid + s];
    __syncthreads();
  }
  const float inv = rsqrtf(red[0] / (float)DI + 1e-5f);
#pragma unroll
  for (int i = 0; i < 4; ++i) {
    const int d = tid + i * 256;
    ybf[(size_t)tok * DI + d] = (bf16_t)(v[i] * inv * nw[d]);
  }
}

// ---- out_proj GEMM + fused MoE combine: out += w_tok * (Y * Wo^T) ----------
// block = 256 threads = 8 waves; each wave: 32(M) x 64(N); N = 512 exactly.
__global__ void k_gemm_outproj(const bf16_t* __restrict__ Yb,
                               const bf16_t* __restrict__ Wo,
                               const float* __restrict__ gw,  // stride NEXP
                               float* __restrict__ Out) {
  const int wv    = threadIdx.x >> 5;
  const int m0    = blockIdx.y << 5;
  const int nbase = wv << 6;
  v8f acc[2][4] = {};
  wave_gemm32<4, DI>(Yb + (size_t)m0 * DI, Wo + (size_t)nbase * DI, acc);

  const int lane = threadIdx.x & 31;
  const int col  = lane & 15;
  const int rb   = (lane >> 4) << 3;
#pragma unroll
  for (int half = 0; half < 2; ++half) {
#pragma unroll
    for (int r = 0; r < 8; ++r) {
      const int row = m0 + half * 16 + rb + r;
      const float w = gw[(size_t)row * NEXP];
      float* o = Out + (size_t)row * DM + nbase + col;
#pragma unroll
      for (int s = 0; s < 4; ++s)
        o[s << 4] += w * acc[half][s][r];
    }
  }
}

// ---- host-side orchestration ------------------------------------------------
extern "C" void kernel_launch(void* const* d_in, const int* in_sizes, int n_in,
                              void* d_out, int out_size, void* d_ws, size_t ws_size,
                              hipStream_t stream) {
  (void)in_sizes; (void)n_in; (void)out_size; (void)ws_size;
  const float* x          = (const float*)d_in[0];
  const float* gate_w     = (const float*)d_in[1];
  const float* gate_b     = (const float*)d_in[2];
  const float* gate_noise = (const float*)d_in[3];
  const float* in_proj_w  = (const float*)d_in[4];
  const float* conv_w     = (const float*)d_in[5];
  const float* conv_b     = (const float*)d_in[6];
  const float* dt_bias    = (const float*)d_in[7];
  const float* A_log      = (const float*)d_in[8];
  const float* Dvec       = (const float*)d_in[9];
  const float* norm_w     = (const float*)d_in[10];
  const float* out_proj_w = (const float*)d_in[11];
  float* out = (float*)d_out;

  // workspace carve-out (256B aligned)
  char* wp = (char*)d_ws;
  auto alloc = [&](size_t bytes) -> char* {
    char* p = wp;
    wp += (bytes + 255) & ~(size_t)255;
    return p;
  };
  float*  maskw = (float*) alloc((size_t)T_TOK * NEXP * sizeof(float));
  bf16_t* x_bf  = (bf16_t*)alloc((size_t)T_TOK * DM * sizeof(bf16_t));
  bf16_t* wi_bf = (bf16_t*)alloc((size_t)NEXP * DIN * DM * sizeof(bf16_t));
  bf16_t* wo_bf = (bf16_t*)alloc((size_t)NEXP * DM * DI * sizeof(bf16_t));
  float*  zx    = (float*) alloc((size_t)T_TOK * DIN * sizeof(float));   // per-expert reuse
  float*  xc    = (float*) alloc((size_t)T_TOK * CDIM * sizeof(float));
  float*  yraw  = (float*) alloc((size_t)T_TOK * DI * sizeof(float));
  bf16_t* ybf   = (bf16_t*)alloc((size_t)T_TOK * DI * sizeof(bf16_t));

  // gating + precision conversion + output clear
  k_gate<<<dim3(T_TOK / 8), dim3(256), 0, stream>>>(x, gate_w, gate_b, gate_noise, maskw);
  {
    const int n0 = T_TOK * DM;
    k_cvt<<<dim3((n0 + 255) / 256), dim3(256), 0, stream>>>(x, x_bf, n0);
    const int n1 = NEXP * DIN * DM;
    k_cvt<<<dim3((n1 + 255) / 256), dim3(256), 0, stream>>>(in_proj_w, wi_bf, n1);
    const int n2 = NEXP * DM * DI;
    k_cvt<<<dim3((n2 + 255) / 256), dim3(256), 0, stream>>>(out_proj_w, wo_bf, n2);
    const int n3 = T_TOK * DM;
    k_zero<<<dim3((n3 + 255) / 256), dim3(256), 0, stream>>>(out, n3);
  }

  for (int e = 0; e < NEXP; ++e) {
    k_gemm_inproj<<<dim3((DIN + 511) / 512, T_TOK / 32), dim3(256), 0, stream>>>(
        x_bf, wi_bf + (size_t)e * DIN * DM, zx);
    k_conv<<<dim3((T_TOK * CDIM + 255) / 256), dim3(256), 0, stream>>>(
        zx, conv_w + (size_t)e * CDIM * 4, conv_b + (size_t)e * CDIM, xc);
    k_scan<<<dim3(B_SZ * NH), dim3(256), 0, stream>>>(
        zx, xc, dt_bias + e * NH, A_log + e * NH, Dvec + e * NH, yraw);
    k_norm<<<dim3(T_TOK), dim3(256), 0, stream>>>(
        yraw, zx, norm_w + (size_t)e * DI, ybf);
    k_gemm_outproj<<<dim3(1, T_TOK / 32), dim3(256), 0, stream>>>(
        ybf, wo_bf + (size_t)e * DM * DI, maskw + e, out);
  }
}
